// LocationSensitiveAttention_33397665694167
// MI455X (gfx1250) — matrix operand order repaired and checked by reading, not compiled, not run on hardware
//
#include <hip/hip_runtime.h>
#include <hip/hip_bf16.h>

// Shapes (fixed by the reference)
#define T_TOK 1024
#define BATCH 128
#define HENC  512
#define AHID  128
#define KFILT 32
#define FLEN  31

typedef __attribute__((ext_vector_type(16))) __bf16 v16bf;
typedef __attribute__((ext_vector_type(2)))  __bf16 v2bf;
typedef __attribute__((ext_vector_type(8)))  float  v8f;
typedef unsigned short u16;

union Frag16 {
  v16bf v;
  v2bf  h[8];
  u16   us[16];
  uint4 q[2];
};

// Native f32 -> bf16 (RNE)
__device__ __forceinline__ __bf16 f2bf16(float x) { return (__bf16)x; }
__device__ __forceinline__ u16 f2bf_bits(float x) {
  return __builtin_bit_cast(u16, (__bf16)x);
}

// Packed pair conversion: one v_cvt_pk_bf16_f32 where available.
__device__ __forceinline__ v2bf pk_bf16(float lo, float hi) {
#if __has_builtin(__builtin_amdgcn_cvt_pk_bf16_f32)
  return __builtin_amdgcn_cvt_pk_bf16_f32(lo, hi);
#else
  v2bf r;
  r[0] = (__bf16)lo;
  r[1] = (__bf16)hi;
  return r;
#endif
}

// Hardware tanh (V_TANH_F32 transcendental) where exposed.
__device__ __forceinline__ float fast_tanh(float x) {
#if __has_builtin(__builtin_amdgcn_tanhf)
  return __builtin_amdgcn_tanhf(x);
#elif __has_builtin(__builtin_amdgcn_tanh_f32)
  return __builtin_amdgcn_tanh_f32(x);
#else
  return tanhf(x);
#endif
}

// ---------------------------------------------------------------------------
// K1a: WcombT[a][f] (bf16, f padded to 32) = sum_k conv_filter[k,f] * W_loc[k,a]
// Fuses the conv filter bank with W_loc: the whole location path collapses to
// a single K=32 WMMA step later.
__global__ void prep_wcomb(const float* __restrict__ cf,
                           const float* __restrict__ wloc,
                           u16* __restrict__ wcombT) {
  int tid = threadIdx.x;          // 256 threads
  int a  = tid >> 1;
  int f0 = (tid & 1) * 16;
  for (int f = f0; f < f0 + 16; ++f) {
    float s = 0.0f;
    if (f < FLEN) {
      #pragma unroll
      for (int k = 0; k < KFILT; ++k)
        s += cf[k * FLEN + f] * wloc[k * AHID + a];
    }
    wcombT[a * 32 + f] = f2bf_bits(s);
  }
}

// K1b: W_kT[a][h] bf16 = W_k[h][a] (transpose so B-fragments are contiguous)
__global__ void prep_wkT(const float* __restrict__ wk, u16* __restrict__ wkT) {
  int idx = blockIdx.x * 256 + threadIdx.x;   // 65536 total
  int a = idx >> 9;
  int h = idx & 511;
  wkT[a * HENC + h] = f2bf_bits(wk[h * AHID + a]);
}

// K1c: proc_query[b][a] fp32 = query[b,:] @ W_q[:,a]
__global__ void prep_pq(const float* __restrict__ query,
                        const float* __restrict__ wq,
                        float* __restrict__ pq) {
  int b = blockIdx.x, a = threadIdx.x;        // 128 blocks x 128 threads
  float s = 0.0f;
  for (int h = 0; h < HENC; ++h)
    s += query[b * HENC + h] * wq[h * AHID + a];
  pq[b * AHID + a] = s;
}

// ---------------------------------------------------------------------------
// K2: fused score kernel. One WG per token t. Each of 8 waves owns a
// 16(b) x 128(a) output tile; acc = WMMA(loc) then 16 K-steps of WMMA over H.
__global__ __launch_bounds__(256) void score_kernel(
    const float* __restrict__ enc,   // [T,B,H]
    const float* __restrict__ prev,  // [T,B]
    const u16*   __restrict__ wkT,   // [A,H] bf16
    const u16*   __restrict__ wcombT,// [A,32] bf16
    const float* __restrict__ pq,    // [B,A]
    const float* __restrict__ vvec,  // [A]
    float* __restrict__ score) {     // [T,B]
  __shared__ u16 ldsW[128 * 136];    // W_k chunk [a][k], padded stride 136

  const int t    = blockIdx.x;
  const int tid  = threadIdx.x;
  const int lane = tid & 31;
  const int wave = tid >> 5;
  const int n    = lane & 15;
  const int half = lane >> 4;
  const int bRow = wave * 16 + n;    // A-matrix row (= batch index)

  v8f acc[8];

  // ---- location term: prevWindow[16b x 32f] x WcombT -> init accumulators
  Frag16 aL;
  #pragma unroll
  for (int i = 0; i < 16; i += 2) {
    // ISA 16-bit A layout: K = (i>>3)*16 + half*8 + (i&7)
    int f0 = ((i >> 3) << 4) + (half << 3) + (i & 7);
    int f1 = f0 + 1;
    int t0 = t - 15 + f0, t1 = t - 15 + f1;
    float v0 = (f0 < FLEN && t0 >= 0 && t0 < T_TOK) ? prev[t0 * BATCH + bRow] : 0.0f;
    float v1 = (f1 < FLEN && t1 >= 0 && t1 < T_TOK) ? prev[t1 * BATCH + bRow] : 0.0f;
    aL.h[i >> 1] = pk_bf16(v0, v1);
  }
  #pragma unroll
  for (int j = 0; j < 8; ++j) {
    Frag16 bL;
    const uint4* bp = (const uint4*)(wcombT + ((j * 16 + n) * 32 + half * 8));
    bL.q[0] = bp[0];
    bL.q[1] = bp[2];
    v8f z = {};
    acc[j] = __builtin_amdgcn_wmma_f32_16x16x32_bf16(
        false, aL.v, false, bL.v, (short)0, z, false, false);
  }

  // ---- main GEMM: enc[t,b,:] @ W_k over H=512, 4 LDS chunks x 4 K-steps
  const float* rowPtr = enc + (size_t)(t * BATCH + bRow) * HENC;
  // prefetch first chunk of this wave's rows into near caches
  __builtin_prefetch(rowPtr, 0, 1);
  __builtin_prefetch(rowPtr + 64, 0, 1);
  for (int c = 0; c < 4; ++c) {
    __syncthreads();
    {
      int a  = tid & 127;
      int kh = (tid >> 7) * 64;
      const uint4* src = (const uint4*)(wkT + a * HENC + c * 128 + kh);
      uint4* dst = (uint4*)(&ldsW[a * 136 + kh]);
      #pragma unroll
      for (int i = 0; i < 8; ++i) dst[i] = src[i];
    }
    // overlap: pull next chunk's 512B of this wave's enc rows toward the WGP
    if (c < 3) {
      __builtin_prefetch(rowPtr + (c + 1) * 128, 0, 1);
      __builtin_prefetch(rowPtr + (c + 1) * 128 + 64, 0, 1);
    }
    __syncthreads();
    #pragma unroll
    for (int s = 0; s < 4; ++s) {
      const int k0 = c * 128 + s * 32;
      Frag16 A;
      const float4 fa0 = *(const float4*)(rowPtr + k0 + half * 8);
      const float4 fa1 = *(const float4*)(rowPtr + k0 + half * 8 + 4);
      const float4 fb0 = *(const float4*)(rowPtr + k0 + 16 + half * 8);
      const float4 fb1 = *(const float4*)(rowPtr + k0 + 16 + half * 8 + 4);
      A.h[0] = pk_bf16(fa0.x, fa0.y);
      A.h[1] = pk_bf16(fa0.z, fa0.w);
      A.h[2] = pk_bf16(fa1.x, fa1.y);
      A.h[3] = pk_bf16(fa1.z, fa1.w);
      A.h[4] = pk_bf16(fb0.x, fb0.y);
      A.h[5] = pk_bf16(fb0.z, fb0.w);
      A.h[6] = pk_bf16(fb1.x, fb1.y);
      A.h[7] = pk_bf16(fb1.z, fb1.w);
      #pragma unroll
      for (int j = 0; j < 8; ++j) {
        Frag16 B;
        const uint4* bp = (const uint4*)(&ldsW[(j * 16 + n) * 136 + s * 32 + half * 8]);
        B.q[0] = bp[0];
        B.q[1] = bp[2];
        acc[j] = __builtin_amdgcn_wmma_f32_16x16x32_bf16(
            false, A.v, false, B.v, (short)0, acc[j], false, false);
      }
    }
  }

  // ---- epilogue: + proc_query, tanh, dot with v, reduce over 16 lanes (a dim)
  float vv[8];
  #pragma unroll
  for (int j = 0; j < 8; ++j) vv[j] = vvec[j * 16 + n];

  #pragma unroll
  for (int e = 0; e < 8; ++e) {
    // D layout: VGPR e holds M = e + 8*half; N = lane&15
    const int bG = wave * 16 + e + half * 8;
    const float* pqRow = pq + bG * AHID;
    float partial = 0.0f;
    #pragma unroll
    for (int j = 0; j < 8; ++j) {
      float s = acc[j][e] + pqRow[j * 16 + n];
      partial += fast_tanh(s) * vv[j];
    }
    partial += __shfl_xor(partial, 1, 32);
    partial += __shfl_xor(partial, 2, 32);
    partial += __shfl_xor(partial, 4, 32);
    partial += __shfl_xor(partial, 8, 32);
    if (n == 0) score[t * BATCH + bG] = partial;
  }
}

// ---------------------------------------------------------------------------
// K3: softmax over T per batch column; writes alignment into d_out slot.
__global__ __launch_bounds__(256) void softmax_kernel(
    const float* __restrict__ score, float* __restrict__ align) {
  __shared__ float redM[8], redS[8];
  int b = blockIdx.x, tid = threadIdx.x;
  int lane = tid & 31, wave = tid >> 5;
  float v[4], m = -1e30f;
  #pragma unroll
  for (int i = 0; i < 4; ++i) {
    v[i] = score[(tid + i * 256) * BATCH + b];
    m = fmaxf(m, v[i]);
  }
  #pragma unroll
  for (int o = 16; o > 0; o >>= 1) m = fmaxf(m, __shfl_xor(m, o, 32));
  if (lane == 0) redM[wave] = m;
  __syncthreads();
  float bm = redM[0];
  #pragma unroll
  for (int w = 1; w < 8; ++w) bm = fmaxf(bm, redM[w]);
  float e[4], s = 0.0f;
  #pragma unroll
  for (int i = 0; i < 4; ++i) { e[i] = __expf(v[i] - bm); s += e[i]; }
  #pragma unroll
  for (int o = 16; o > 0; o >>= 1) s += __shfl_xor(s, o, 32);
  if (lane == 0) redS[wave] = s;
  __syncthreads();
  float bs = 0.0f;
  #pragma unroll
  for (int w = 0; w < 8; ++w) bs += redS[w];
  float inv = 1.0f / bs;
  #pragma unroll
  for (int i = 0; i < 4; ++i) align[(tid + i * 256) * BATCH + b] = e[i] * inv;
}

// ---------------------------------------------------------------------------
// K4: split-T context partials (no atomics -> deterministic)
__global__ __launch_bounds__(256) void ctx_partial(
    const float* __restrict__ enc, const float* __restrict__ align,
    float* __restrict__ partial) {
  int b = blockIdx.x & 127;
  int c = blockIdx.x >> 7;     // 8 T-chunks of 128
  int tid = threadIdx.x;       // float2 per thread covers H=512
  float2 acc = make_float2(0.f, 0.f);
  #pragma unroll 4
  for (int t = c * 128; t < c * 128 + 128; ++t) {
    float a = align[t * BATCH + b];
    const float2* row = (const float2*)(enc + (size_t)(t * BATCH + b) * HENC);
    float2 ev = row[tid];
    acc.x += a * ev.x;
    acc.y += a * ev.y;
  }
  ((float2*)(partial + (size_t)(c * BATCH + b) * HENC))[tid] = acc;
}

// K5: reduce 8 partials -> context
__global__ __launch_bounds__(256) void ctx_reduce(
    const float* __restrict__ partial, float* __restrict__ ctx) {
  int b = blockIdx.x, tid = threadIdx.x;
  float2 s = make_float2(0.f, 0.f);
  #pragma unroll
  for (int c = 0; c < 8; ++c) {
    float2 v = ((const float2*)(partial + (size_t)(c * BATCH + b) * HENC))[tid];
    s.x += v.x; s.y += v.y;
  }
  ((float2*)(ctx + (size_t)b * HENC))[tid] = s;
}

// ---------------------------------------------------------------------------
extern "C" void kernel_launch(void* const* d_in, const int* in_sizes, int n_in,
                              void* d_out, int out_size, void* d_ws, size_t ws_size,
                              hipStream_t stream) {
  const float* enc   = (const float*)d_in[0];  // [T,B,H]
  const float* query = (const float*)d_in[1];  // [B,H]
  const float* prev  = (const float*)d_in[2];  // [T,B]
  const float* convf = (const float*)d_in[3];  // [K,1,F]
  const float* wloc  = (const float*)d_in[4];  // [K,A]
  const float* wq    = (const float*)d_in[5];  // [H,A]
  const float* wk    = (const float*)d_in[6];  // [H,A]
  const float* vvec  = (const float*)d_in[7];  // [A]

  float* out_ctx   = (float*)d_out;                    // [B,H]
  float* out_align = (float*)d_out + BATCH * HENC;     // [T,B]

  // workspace layout (bytes): ~2.7 MB total
  char* ws = (char*)d_ws;
  u16*   wkT     = (u16*)(ws + 0);        // 128*512*2  = 131072
  u16*   wcombT  = (u16*)(ws + 131072);   // 128*32*2   = 8192
  float* pq      = (float*)(ws + 139264); // 128*128*4  = 65536
  float* score   = (float*)(ws + 204800); // 1024*128*4 = 524288
  float* partial = (float*)(ws + 729088); // 8*128*512*4 = 2097152

  prep_wcomb<<<1, 256, 0, stream>>>(convf, wloc, wcombT);
  prep_wkT<<<256, 256, 0, stream>>>(wk, wkT);
  prep_pq<<<BATCH, AHID, 0, stream>>>(query, wq, pq);
  score_kernel<<<T_TOK, 256, 0, stream>>>(enc, prev, wkT, wcombT, pq, vvec, score);
  softmax_kernel<<<BATCH, 256, 0, stream>>>(score, out_align);
  ctx_partial<<<BATCH * 8, 256, 0, stream>>>(enc, out_align, partial);
  ctx_reduce<<<BATCH, 256, 0, stream>>>(partial, out_ctx);
}
